// ALiBiAttention_20933670600911
// MI455X (gfx1250) — matrix operand order repaired and checked by reading, not compiled, not run on hardware
//
#include <hip/hip_runtime.h>
#include <hip/hip_bf16.h>
#include <math.h>

typedef __attribute__((ext_vector_type(16))) __bf16 v16bf;
typedef __attribute__((ext_vector_type(8)))  float  v8f;
typedef __attribute__((ext_vector_type(4)))  unsigned int v4u;
typedef __attribute__((ext_vector_type(4)))  int    v4i;
typedef __attribute__((ext_vector_type(8)))  int    v8i;

static constexpr int D_MODEL  = 1024;
static constexpr int N_HEADS  = 16;
static constexpr int HEAD_DIM = 64;
static constexpr int BATCH    = 2;
static constexpr int SEQ      = 2048;
static constexpr int MTOT     = BATCH * SEQ;   // 4096 rows
static constexpr int KTILES   = SEQ / 64;      // 32 key tiles

// ---------------------------------------------------------------- utilities

__device__ __forceinline__ unsigned short f32_to_bf16(float f) {
  unsigned int u = __float_as_uint(f);
  u += 0x7fffu + ((u >> 16) & 1u);          // round-to-nearest-even
  return (unsigned short)(u >> 16);
}

union BFrag { v16bf v; unsigned int u[8]; };

// A-matrix 16x32 bf16 fragment (ISA 7.12.2): lane = row(M),
// kb = 16*(j/4) + 8*(lane>=16) + 2*(j%4). Source is [row][k] row-major.
__device__ __forceinline__ v16bf load_frag_a(const unsigned short* p, int stride,
                                             int lane, int k_off) {
  BFrag f;
  int r  = lane & 15;
  int kh = (lane >> 4) << 3;
#pragma unroll
  for (int j = 0; j < 8; ++j) {
    int kb = ((j >> 2) << 4) + kh + ((j & 3) << 1) + k_off;
    f.u[j] = *(const unsigned int*)(p + r * stride + kb);
  }
  return f.v;
}

// B-matrix 32x16 bf16 fragment: lane = column(N), kb = 16*(j/4) + 8*(lane>=16) + 2*(j%4).
// Source must be [n][k] row-major (i.e. B transposed / N-major).
__device__ __forceinline__ v16bf load_frag_b(const unsigned short* p, int stride,
                                             int lane, int k_off) {
  BFrag f;
  int r  = lane & 15;
  int kh = (lane >> 4) << 3;
#pragma unroll
  for (int j = 0; j < 8; ++j) {
    int kb = ((j >> 2) << 4) + kh + ((j & 3) << 1) + k_off;
    f.u[j] = *(const unsigned int*)(p + r * stride + kb);
  }
  return f.v;
}

__device__ __forceinline__ v8f wmma_bf16(v16bf a, v16bf b, v8f c) {
  return __builtin_amdgcn_wmma_f32_16x16x32_bf16(false, a, false, b,
                                                 (short)0, c, false, false);
}

__device__ __forceinline__ void store_out(float* p, float v) { *p = v; }
__device__ __forceinline__ void store_out(unsigned short* p, float v) { *p = f32_to_bf16(v); }

// ---- Tensor Data Mover: DMA a [tile_h x tile_w] bf16 tile (row stride
// row_stride_elems) from global memory into LDS at lds_off. 2-D descriptor,
// groups 2/3 zero (tile_dim2 == 0 -> higher dims unused).
__device__ __forceinline__ void tdm_load_tile_bf16(unsigned int lds_off,
                                                   const unsigned short* gsrc,
                                                   unsigned int tile_w,
                                                   unsigned int tile_h,
                                                   unsigned int row_stride_elems) {
  unsigned long long ga = (unsigned long long)(size_t)gsrc;
  v4u g0;
  g0[0] = 1u;                                       // count=1 (valid), user mode
  g0[1] = lds_off;                                  // lds_addr  (bits 63:32)
  g0[2] = (unsigned int)(ga & 0xffffffffu);         // global_addr[31:0]
  g0[3] = (unsigned int)((ga >> 32) & 0x01ffffffu)  // global_addr[56:32]
          | (2u << 30);                             // type = 2 ("image")
  unsigned int td0 = row_stride_elems;              // tensor_dim0 (OOB bound)
  unsigned int td1 = 1u << 20;                      // tensor_dim1: generous
  v8i g1;
  g1[0] = (int)(1u << 16);                          // data_size=1 (2B); no mask
  g1[1] = (int)((td0 & 0xffffu) << 16);             // tensor_dim0[15:0]
  g1[2] = (int)((td0 >> 16) | ((td1 & 0xffffu) << 16));
  g1[3] = (int)((td1 >> 16) | (tile_w << 16));      // tile_dim0
  g1[4] = (int)tile_h;                              // tile_dim1; tile_dim2=0
  g1[5] = (int)row_stride_elems;                    // tensor_dim0_stride[31:0]
  g1[6] = 0;                                        // stride0 hi / stride1 lo
  g1[7] = 0;
  v4i gz = {0, 0, 0, 0};
#if defined(__clang_major__) && (__clang_major__ >= 23)
  v8i gz8 = {0, 0, 0, 0, 0, 0, 0, 0};
  __builtin_amdgcn_tensor_load_to_lds(g0, g1, gz, gz, gz8, 0);
#else
  __builtin_amdgcn_tensor_load_to_lds(g0, g1, gz, gz, 0);
#endif
}

// ------------------------------------------------------- GEMM: C = a*(A*W+b)
// A: [M,K] fp32 row-major, W: [K,N] fp32 row-major, bias: [N].
// Block 256 thr = 8 waves (4x2), tile 128x128, K-step 32.

template <typename OutT>
__global__ __launch_bounds__(256)
void gemm_bias_kernel(const float* __restrict__ A, const float* __restrict__ W,
                      const float* __restrict__ bias, OutT* __restrict__ C,
                      int M, int N, int K, float alpha) {
  __shared__ unsigned short As[128 * 32];   // [m][k]
  __shared__ unsigned short Bs[128 * 32];   // [n][k]  (W tile transposed)

  const int tid  = threadIdx.x;
  const int lane = tid & 31;
  const int w    = tid >> 5;
  const int wm   = w & 3;      // 0..3  -> 32-row slab
  const int wn   = w >> 2;     // 0..1  -> 64-col slab
  const int gm0  = blockIdx.y * 128;
  const int gn0  = blockIdx.x * 128;

  v8f zf = {0.f, 0.f, 0.f, 0.f, 0.f, 0.f, 0.f, 0.f};
  v8f acc[2][4];
#pragma unroll
  for (int i = 0; i < 2; ++i)
#pragma unroll
    for (int j = 0; j < 4; ++j) acc[i][j] = zf;

  for (int kk = 0; kk < K; kk += 32) {
    // stage A tile 128x32 (fp32 -> bf16)
#pragma unroll
    for (int i = 0; i < 4; ++i) {
      int r = (tid >> 3) + i * 32;
      int c = (tid & 7) * 4;
      const float* src = A + (size_t)(gm0 + r) * K + kk + c;
      float4 d = *(const float4*)src;
      if (i == 0 && kk + 32 < K) __builtin_prefetch(src + 32, 0, 1);
      unsigned int p0 = f32_to_bf16(d.x) | ((unsigned int)f32_to_bf16(d.y) << 16);
      unsigned int p1 = f32_to_bf16(d.z) | ((unsigned int)f32_to_bf16(d.w) << 16);
      unsigned int* dst = (unsigned int*)&As[r * 32 + c];
      dst[0] = p0; dst[1] = p1;
    }
    // stage W tile 32x128, transposed into Bs[n][k]
#pragma unroll
    for (int i = 0; i < 4; ++i) {
      int kr = (tid >> 5) + i * 8;
      int n  = (tid & 31) * 4;
      const float* src = W + (size_t)(kk + kr) * N + gn0 + n;
      float4 d = *(const float4*)src;
      if (i == 0 && kk + 32 < K) __builtin_prefetch(src + (size_t)32 * N, 0, 1);
      Bs[(n + 0) * 32 + kr] = f32_to_bf16(d.x);
      Bs[(n + 1) * 32 + kr] = f32_to_bf16(d.y);
      Bs[(n + 2) * 32 + kr] = f32_to_bf16(d.z);
      Bs[(n + 3) * 32 + kr] = f32_to_bf16(d.w);
    }
    __syncthreads();

    v16bf af[2], bf[4];
#pragma unroll
    for (int i = 0; i < 2; ++i)
      af[i] = load_frag_a(&As[(wm * 32 + i * 16) * 32], 32, lane, 0);
#pragma unroll
    for (int j = 0; j < 4; ++j)
      bf[j] = load_frag_b(&Bs[(wn * 64 + j * 16) * 32], 32, lane, 0);
#pragma unroll
    for (int i = 0; i < 2; ++i)
#pragma unroll
      for (int j = 0; j < 4; ++j)
        acc[i][j] = wmma_bf16(af[i], bf[j], acc[i][j]);
    __syncthreads();
  }

  // epilogue: C layout -> lane = column, vgpr v -> row v + 8*(lane>=16)
#pragma unroll
  for (int i = 0; i < 2; ++i)
#pragma unroll
    for (int j = 0; j < 4; ++j) {
      int n = gn0 + wn * 64 + j * 16 + (lane & 15);
      float bn = bias[n];
      int mbase = gm0 + wm * 32 + i * 16 + ((lane >> 4) << 3);
#pragma unroll
      for (int v = 0; v < 8; ++v) {
        float val = alpha * (acc[i][j][v] + bn);
        store_out(C + (size_t)(mbase + v) * N + n, val);
      }
    }
}

// -------------------------------------------- flash attention with ALiBi
// Q,K,V: bf16 [B*S][D_MODEL], head h at cols h*64..h*64+63. Q pre-scaled by
// Hd^-0.5. 4 waves / block; block owns 64 q-rows of one (b,h); each wave owns
// 16 q-rows. K tiles DMA'd by the Tensor Data Mover into double-buffered LDS
// (TENSORcnt), V tiles manually transposed; both pipelined one tile ahead.

__global__ __launch_bounds__(128)
void attn_kernel(const unsigned short* __restrict__ Qb,
                 const unsigned short* __restrict__ Kb,
                 const unsigned short* __restrict__ Vb,
                 float* __restrict__ O) {
  __shared__ unsigned short Kt[2][64 * 64];   // [key][d] : B-frag layout, TDM dest
  __shared__ unsigned short Vt[2][64 * 64];   // [d][key] : B-frag layout for P*V
  __shared__ unsigned short Ps[4][16 * 64];   // per-wave P scratch [m][key]

  const int tid  = threadIdx.x;
  const int lane = tid & 31;
  const int wid  = tid >> 5;
  const int b    = blockIdx.y >> 4;
  const int h    = blockIdx.y & 15;
  const int q0   = blockIdx.x * 64;

  // Q fragments straight from global (bf16, A layout)
  const unsigned short* Qp =
      Qb + (size_t)(b * SEQ + q0 + wid * 16) * D_MODEL + h * HEAD_DIM;
  v16bf qf[2];
  qf[0] = load_frag_a(Qp, D_MODEL, lane, 0);
  qf[1] = load_frag_a(Qp, D_MODEL, lane, 32);

  const float slope = exp2f(-(float)h / (float)N_HEADS);

  float mrow[8], lrow[8];
#pragma unroll
  for (int v = 0; v < 8; ++v) { mrow[v] = -1e30f; lrow[v] = 0.0f; }
  v8f zf = {0.f, 0.f, 0.f, 0.f, 0.f, 0.f, 0.f, 0.f};
  v8f oacc[4];
#pragma unroll
  for (int j = 0; j < 4; ++j) oacc[j] = zf;

  const int key_t = tid >> 1;   // 0..63
  const int half  = tid & 1;    // which 32-d half
  const size_t head_base = (size_t)(b * SEQ) * D_MODEL + h * HEAD_DIM;

  // ---- prologue: bring in tile 0 (K via TDM, V via manual transpose)
  if (wid == 0)
    tdm_load_tile_bf16((unsigned int)(size_t)&Kt[0][0], Kb + head_base,
                       HEAD_DIM, 64, D_MODEL);
  {
    const uint4* Vsrc =
        (const uint4*)(Vb + head_base + (size_t)key_t * D_MODEL + half * 32);
#pragma unroll
    for (int i = 0; i < 4; ++i) {
      union { uint4 q; unsigned short s[8]; } uv;
      uv.q = Vsrc[i];
#pragma unroll
      for (int e = 0; e < 8; ++e)
        Vt[0][(half * 32 + i * 8 + e) * 64 + key_t] = uv.s[e];
    }
  }

  for (int it = 0; it < KTILES; ++it) {
    const int buf = it & 1;
    const int kt0 = it * 64;

    __builtin_amdgcn_s_wait_tensorcnt(0);   // K tile DMA complete (wave 0)
    __syncthreads();                        // tile `buf` visible; `buf^1` free

    // ---- pipeline: start tile it+1 while computing tile it
    if (it + 1 < KTILES) {
      const size_t nbase = head_base + (size_t)(kt0 + 64) * D_MODEL;
      if (wid == 0)
        tdm_load_tile_bf16((unsigned int)(size_t)&Kt[buf ^ 1][0], Kb + nbase,
                           HEAD_DIM, 64, D_MODEL);
      const uint4* Vsrc =
          (const uint4*)(Vb + nbase + (size_t)key_t * D_MODEL + half * 32);
#pragma unroll
      for (int i = 0; i < 4; ++i) {
        union { uint4 q; unsigned short s[8]; } uv;
        uv.q = Vsrc[i];
#pragma unroll
        for (int e = 0; e < 8; ++e)
          Vt[buf ^ 1][(half * 32 + i * 8 + e) * 64 + key_t] = uv.s[e];
      }
    }

    // ---- S = Q * K^T   (4 n-tiles x 2 k-chunks = 8 WMMA)
    v8f sacc[4];
#pragma unroll
    for (int j = 0; j < 4; ++j) sacc[j] = zf;
#pragma unroll
    for (int j = 0; j < 4; ++j)
#pragma unroll
      for (int kc = 0; kc < 2; ++kc) {
        v16bf kfrag = load_frag_b(&Kt[buf][(j * 16) * 64], 64, lane, kc * 32);
        sacc[j] = wmma_bf16(qf[kc], kfrag, sacc[j]);
      }

    // ---- ALiBi bias: slope * (key - q)   (scale already folded into Q)
    const float qbase = (float)(q0 + wid * 16 + ((lane >> 4) << 3));
#pragma unroll
    for (int j = 0; j < 4; ++j) {
      float keyf = (float)(kt0 + j * 16 + (lane & 15));
#pragma unroll
      for (int v = 0; v < 8; ++v)
        sacc[j][v] += slope * (keyf - (qbase + (float)v));
    }

    // ---- online softmax (row lives in a 16-lane half)
    float corr[8];
#pragma unroll
    for (int v = 0; v < 8; ++v) {
      float t = fmaxf(fmaxf(sacc[0][v], sacc[1][v]), fmaxf(sacc[2][v], sacc[3][v]));
#pragma unroll
      for (int m = 1; m < 16; m <<= 1) t = fmaxf(t, __shfl_xor(t, m, 32));
      float mnew = fmaxf(mrow[v], t);
      corr[v] = __expf(mrow[v] - mnew);
      mrow[v] = mnew;
    }
#pragma unroll
    for (int v = 0; v < 8; ++v) {
      float rs = 0.0f;
#pragma unroll
      for (int j = 0; j < 4; ++j) {
        float p = __expf(sacc[j][v] - mrow[v]);
        sacc[j][v] = p;
        rs += p;
      }
#pragma unroll
      for (int m = 1; m < 16; m <<= 1) rs += __shfl_xor(rs, m, 32);
      lrow[v] = lrow[v] * corr[v] + rs;
#pragma unroll
      for (int j = 0; j < 4; ++j) oacc[j][v] *= corr[v];
    }

    // ---- P: C layout -> LDS [m][key] -> A fragments
    unsigned short* myP = &Ps[wid][0];
    {
      int rb = (lane >> 4) << 3;
      int cb = lane & 15;
#pragma unroll
      for (int j = 0; j < 4; ++j)
#pragma unroll
        for (int v = 0; v < 8; ++v)
          myP[(rb + v) * 64 + j * 16 + cb] = f32_to_bf16(sacc[j][v]);
    }
    __syncthreads();   // order P writes vs cross-lane P reads

    v16bf pf[2];
    pf[0] = load_frag_a(myP, 64, lane, 0);
    pf[1] = load_frag_a(myP, 64, lane, 32);

    // ---- O += P * V   (8 WMMA)
#pragma unroll
    for (int j = 0; j < 4; ++j)
#pragma unroll
      for (int kc = 0; kc < 2; ++kc) {
        v16bf vfrag = load_frag_b(&Vt[buf][(j * 16) * 64], 64, lane, kc * 32);
        oacc[j] = wmma_bf16(pf[kc], vfrag, oacc[j]);
      }
  }

  // ---- normalize and store fp32 attention output [B*S][D_MODEL]
  int rb = (lane >> 4) << 3;
  int cb = lane & 15;
#pragma unroll
  for (int j = 0; j < 4; ++j)
#pragma unroll
    for (int v = 0; v < 8; ++v) {
      int row = q0 + wid * 16 + rb + v;
      int col = h * HEAD_DIM + j * 16 + cb;
      O[(size_t)(b * SEQ + row) * D_MODEL + col] = oacc[j][v] / lrow[v];
    }
}

// ----------------------------------------------------------------- launcher

extern "C" void kernel_launch(void* const* d_in, const int* in_sizes, int n_in,
                              void* d_out, int out_size, void* d_ws, size_t ws_size,
                              hipStream_t stream) {
  const float* x  = (const float*)d_in[0];
  const float* Wq = (const float*)d_in[1];
  const float* bq = (const float*)d_in[2];
  const float* Wk = (const float*)d_in[3];
  const float* bk = (const float*)d_in[4];
  const float* Wv = (const float*)d_in[5];
  const float* bv = (const float*)d_in[6];
  const float* Wo = (const float*)d_in[7];
  const float* bo = (const float*)d_in[8];
  float* out = (float*)d_out;

  // workspace: Q,K,V bf16 (8 MB each) + attention output fp32 (16 MB)
  unsigned short* Qb = (unsigned short*)d_ws;
  unsigned short* Kb = Qb + (size_t)MTOT * D_MODEL;
  unsigned short* Vb = Kb + (size_t)MTOT * D_MODEL;
  float* attn = (float*)(Vb + (size_t)MTOT * D_MODEL);

  dim3 gg(D_MODEL / 128, MTOT / 128);   // (8, 32) blocks of 256
  const float scale = 0.125f;           // HEAD_DIM^-0.5, folded into Q
  gemm_bias_kernel<unsigned short><<<gg, 256, 0, stream>>>(x, Wq, bq, Qb, MTOT, D_MODEL, D_MODEL, scale);
  gemm_bias_kernel<unsigned short><<<gg, 256, 0, stream>>>(x, Wk, bk, Kb, MTOT, D_MODEL, D_MODEL, 1.0f);
  gemm_bias_kernel<unsigned short><<<gg, 256, 0, stream>>>(x, Wv, bv, Vb, MTOT, D_MODEL, D_MODEL, 1.0f);

  dim3 ga(SEQ / 64, BATCH * N_HEADS);   // (32, 32) blocks of 128
  attn_kernel<<<ga, 128, 0, stream>>>(Qb, Kb, Vb, attn);

  gemm_bias_kernel<float><<<gg, 256, 0, stream>>>(attn, Wo, bo, out, MTOT, D_MODEL, D_MODEL, 1.0f);
}
